// CifPredictor_90254442758599
// MI455X (gfx1250) — matrix operand order repaired and checked by reading, not compiled, not run on hardware
//
#include <hip/hip_runtime.h>
#include <hip/hip_bf16.h>
#include <stdint.h>

// ---- problem constants (fixed by the reference) ----
#define BB   32
#define TT   1000
#define DD   512
#define UU   5
#define KC   3
#define TT1  1001          // T+1
#define TP   1024          // padded M rows per batch for 128-row tiles
#define APAD_T 1026        // TP + 2 halo rows
#define TAILC 0.45f

typedef __bf16 v8bf  __attribute__((ext_vector_type(8)));
typedef __bf16 v16bf __attribute__((ext_vector_type(16)));
typedef float  v8f   __attribute__((ext_vector_type(8)));
typedef unsigned int u32x4 __attribute__((ext_vector_type(4)));
typedef int          i32x8 __attribute__((ext_vector_type(8)));

union ABfrag { v16bf v; v8bf h[2]; };

static __device__ inline unsigned short f2bf(float f) {
    unsigned int u = __float_as_uint(f);
    unsigned int r = u + 0x7FFFu + ((u >> 16) & 1u);   // round-to-nearest-even
    return (unsigned short)(r >> 16);
}

static __device__ inline float sigmoidf(float x) {
    return 1.0f / (1.0f + __expf(-x));
}

// ---------------------------------------------------------------------------
// TDM: issue a 2D tile DMA global->LDS via a hand-packed Tensor DMA Descriptor
// (CDNA5 ISA). Tracked with TENSORcnt.
//   group0: count=1 | lds_addr | global_addr | type=2
//   group1: data_size=2B, big tensor dims (no OOB clip), tile dims, row stride
// ---------------------------------------------------------------------------
static __device__ inline void tdm_load_2d(unsigned ldsAddr, const void* gptr,
                                          unsigned tileCols, unsigned tileRows,
                                          unsigned rowStrideElems)
{
    unsigned long ga = (unsigned long)(uintptr_t)gptr;
    u32x4 g0;
    g0.x = 1u;                                                  // count=1, user load
    g0.y = ldsAddr;                                             // LDS byte address
    g0.z = (unsigned)ga;                                        // global_addr[31:0]
    g0.w = (unsigned)((ga >> 32) & 0x01FFFFFFu) | (2u << 30);   // addr[56:32]|type=2
    const unsigned td0 = 1u << 20, td1 = 1u << 20;              // huge tensor dims
    const unsigned long st0 = rowStrideElems;                   // dim0 stride (elems)
    i32x8 g1;
    g1[0] = (int)(1u << 16);                                    // data_size=1 (2B)
    g1[1] = (int)((td0 & 0xFFFFu) << 16);                       // atomic_addr=0|dim0.lo
    g1[2] = (int)((td0 >> 16) | ((td1 & 0xFFFFu) << 16));       // dim0.hi|dim1.lo
    g1[3] = (int)((td1 >> 16) | (tileCols << 16));              // dim1.hi|tile_dim0
    g1[4] = (int)tileRows;                                      // tile_dim1|tile_dim2=0
    g1[5] = (int)(unsigned)(st0 & 0xFFFFFFFFu);                 // dim0_stride lo
    g1[6] = (int)(unsigned)(st0 >> 32);                         // dim0_stride hi
    g1[7] = 0;
    asm volatile("tensor_load_to_lds %0, %1" :: "s"(g0), "s"(g1) : "memory");
}

// ---------------------------------------------------------------------------
// 1) Pad + cast encoder_out -> Apad bf16 [B][APAD_T][D]; row r holds enc[r-1],
//    zeros for r==0 and r>T (covers conv halo and M-tile overrun to t=1023).
// ---------------------------------------------------------------------------
__global__ __launch_bounds__(256) void prep_apad_kernel(
    const float* __restrict__ enc, unsigned short* __restrict__ Apad)
{
    long idx = (long)blockIdx.x * 256 + threadIdx.x;
    if (idx >= (long)BB * APAD_T * DD) return;
    int d = (int)(idx & (DD - 1));
    long rem = idx >> 9;
    int r = (int)(rem % APAD_T);
    int b = (int)(rem / APAD_T);
    int t = r - 1;
    float val = (t >= 0 && t < TT) ? enc[(((long)b * TT) + t) * DD + d] : 0.0f;
    Apad[idx] = f2bf(val);
}

// ---------------------------------------------------------------------------
// 2) Repack conv_w [O][I][K] -> Wc bf16 [k][o][i] (B-fragment contiguous in i)
// ---------------------------------------------------------------------------
__global__ __launch_bounds__(256) void prep_wc_kernel(
    const float* __restrict__ conv_w, unsigned short* __restrict__ Wc)
{
    int idx = blockIdx.x * 256 + threadIdx.x;
    if (idx >= KC * DD * DD) return;
    int i = idx & (DD - 1);
    int o = (idx >> 9) & (DD - 1);
    int k = idx >> 18;
    Wc[idx] = f2bf(conv_w[(((long)o << 9) + i) * KC + k]);
}

// ---------------------------------------------------------------------------
// 3) Fuse upsampler with w2: v[i][j] = sum_o up_w[i,o,j] * w2[o]; v[2560] = up_b.w2
// ---------------------------------------------------------------------------
__global__ __launch_bounds__(256) void prep_v_kernel(
    const float* __restrict__ up_w, const float* __restrict__ up_b,
    const float* __restrict__ w2, float* __restrict__ vmat)
{
    int idx = blockIdx.x * 256 + threadIdx.x;
    if (idx < DD * UU) {
        int i = idx / UU, j = idx % UU;
        float s = 0.0f;
        for (int o = 0; o < DD; ++o)
            s += up_w[(((long)i << 9) + o) * UU + j] * w2[o];
        vmat[idx] = s;
    } else if (idx == DD * UU) {
        float s = 0.0f;
        for (int o = 0; o < DD; ++o) s += up_b[o] * w2[o];
        vmat[idx] = s;
    }
}

// ---------------------------------------------------------------------------
// 4) Conv1d as WMMA GEMM with TDM->LDS double-buffered staging.
//    out[b,t,o] = relu(bias[o] + sum_k sum_i Apad[b,t+k,i]*Wc[k,o,i])
//    Block: 256 thr = 8 waves as 4(M)x2(N); wave tile 32x32; block tile 128x64.
//    K pipeline: 3 conv taps x 8 steps of 64; wave0 issues 2 tensor_load_to_lds
//    per step; all waves s_wait_tensorcnt + barrier; fragments via ds_load.
// ---------------------------------------------------------------------------
__global__ __launch_bounds__(256) void conv_gemm_wmma_kernel(
    const unsigned short* __restrict__ Apad,
    const unsigned short* __restrict__ Wc,
    const float* __restrict__ conv_b,
    float* __restrict__ outF)
{
    __shared__ unsigned short lA[2][128 * 64];   // 2 x 16 KB
    __shared__ unsigned short lB[2][64 * 64];    // 2 x  8 KB

    const int lane   = threadIdx.x & 31;
    const int wave   = threadIdx.x >> 5;
    const int wm     = wave >> 1;          // 0..3
    const int wn     = wave & 1;           // 0..1
    const int b      = blockIdx.z;
    const int t0blk  = blockIdx.y * 128;
    const int n0blk  = blockIdx.x * 64;
    const int laneLo = lane & 15;
    const int laneHi = lane >> 4;          // 0 or 1

    v8f acc[2][2];
#pragma unroll
    for (int i = 0; i < 2; ++i)
#pragma unroll
        for (int j = 0; j < 2; ++j) acc[i][j] = (v8f)0.0f;

    const long arow_base = (long)b * APAD_T;
    const int STEPS = KC * 8;   // 24 K-steps of 64

    auto issue = [&](int s) {
        const int kc   = s >> 3;
        const int kk   = (s & 7) << 6;
        const int bufi = s & 1;
        const unsigned short* ga =
            Apad + ((arow_base + t0blk + kc) << 9) + kk;          // A tile origin
        const unsigned short* gb =
            Wc + ((((long)kc << 9) + n0blk) << 9) + kk;           // B tile origin
        tdm_load_2d((unsigned)(uintptr_t)&lA[bufi][0], ga, 64, 128, DD);
        tdm_load_2d((unsigned)(uintptr_t)&lB[bufi][0], gb, 64, 64,  DD);
    };

    if (wave == 0) { issue(0); issue(1); }

    for (int s = 0; s < STEPS; ++s) {
        if (s < STEPS - 1) __builtin_amdgcn_s_wait_tensorcnt(2);   // step s done
        else               __builtin_amdgcn_s_wait_tensorcnt(0);   // drain
        __syncthreads();

        const unsigned short* As = &lA[s & 1][0];
        const unsigned short* Bs = &lB[s & 1][0];

#pragma unroll
        for (int kch = 0; kch < 2; ++kch) {   // two 32-wide WMMA K-chunks
            // A frag (16-bit A 16x32): row=laneLo, K at kch*32 + laneHi*8 (+16)
            const unsigned short* aB0 =
                As + ((wm * 32 + 0 * 16 + laneLo) << 6) + kch * 32 + (laneHi << 3);
            const unsigned short* aB1 =
                As + ((wm * 32 + 1 * 16 + laneLo) << 6) + kch * 32 + (laneHi << 3);
            // B frag (16-bit B 32x16): col=laneLo, K at kch*32 + laneHi*8 (+16)
            const unsigned short* bB0 =
                Bs + ((wn * 32 + 0 * 16 + laneLo) << 6) + kch * 32 + (laneHi << 3);
            const unsigned short* bB1 =
                Bs + ((wn * 32 + 1 * 16 + laneLo) << 6) + kch * 32 + (laneHi << 3);

            ABfrag a0, a1, f0, f1;
            a0.h[0] = *(const v8bf*)(aB0);      a0.h[1] = *(const v8bf*)(aB0 + 16);
            a1.h[0] = *(const v8bf*)(aB1);      a1.h[1] = *(const v8bf*)(aB1 + 16);
            f0.h[0] = *(const v8bf*)(bB0);      f0.h[1] = *(const v8bf*)(bB0 + 16);
            f1.h[0] = *(const v8bf*)(bB1);      f1.h[1] = *(const v8bf*)(bB1 + 16);

            acc[0][0] = __builtin_amdgcn_wmma_f32_16x16x32_bf16(
                false, a0.v, false, f0.v, (short)0, acc[0][0], false, false);
            acc[0][1] = __builtin_amdgcn_wmma_f32_16x16x32_bf16(
                false, a0.v, false, f1.v, (short)0, acc[0][1], false, false);
            acc[1][0] = __builtin_amdgcn_wmma_f32_16x16x32_bf16(
                false, a1.v, false, f0.v, (short)0, acc[1][0], false, false);
            acc[1][1] = __builtin_amdgcn_wmma_f32_16x16x32_bf16(
                false, a1.v, false, f1.v, (short)0, acc[1][1], false, false);
        }

        __syncthreads();                       // buf[s&1] free for reuse
        if (wave == 0 && s + 2 < STEPS) issue(s + 2);
    }

    // Epilogue: bias + ReLU, predicated store (only t < T valid).
    // C frag layout: VGPR v -> M = v + laneHi*8 ; N = laneLo.
    const int t0 = t0blk + wm * 32;
    const int n0 = n0blk + wn * 32;
#pragma unroll
    for (int j = 0; j < 2; ++j) {
        const int c = n0 + j * 16 + laneLo;
        const float bias = conv_b[c];
#pragma unroll
        for (int i = 0; i < 2; ++i) {
#pragma unroll
            for (int v = 0; v < 8; ++v) {
                int t = t0 + i * 16 + v + laneHi * 8;
                if (t < TT) {
                    float x = acc[i][j][v] + bias;
                    outF[((((long)b << 10) + t) << 9) + c] = x > 0.0f ? x : 0.0f;
                }
            }
        }
    }
}

// ---------------------------------------------------------------------------
// 5) Pointwise/GEMV: one wave per (b,t): alpha row (-> alphas_t with tail-add)
//    and fused-upsample alphas2 partial sums. t==1000 writes only the tail term.
// ---------------------------------------------------------------------------
__global__ __launch_bounds__(256) void pointwise_kernel(
    const float* __restrict__ outF, const float* __restrict__ mask,
    const float* __restrict__ vmat, const float* __restrict__ w1,
    const float* __restrict__ b1, const float* __restrict__ b2,
    float* __restrict__ alphas_t, float* __restrict__ a2partial)
{
    const int wave = threadIdx.x >> 5;
    const int lane = threadIdx.x & 31;
    const int row  = blockIdx.x * 8 + wave;
    if (row >= BB * TT1) return;
    const int b = row / TT1;
    const int t = row % TT1;

    const float mcur  = (t < TT) ? mask[b * TT + t] : 0.0f;
    const float mprev = (t == 0) ? 1.0f : mask[b * TT + t - 1];

    if (t == TT) {  // tail slot: alphas contribution is 0, only tail_add
        if (lane == 0) alphas_t[b * TT1 + TT] = TAILC * mprev;
        return;
    }

    const float* xrow = outF + ((((long)b << 10) + t) << 9);
    float a1s = 0.0f, z0 = 0.0f, z1 = 0.0f, z2 = 0.0f, z3 = 0.0f, z4 = 0.0f;
    for (int i = lane; i < DD; i += 32) {
        float x = xrow[i];
        a1s += x * w1[i];
        const float* vp = vmat + i * UU;
        z0 += x * vp[0]; z1 += x * vp[1]; z2 += x * vp[2];
        z3 += x * vp[3]; z4 += x * vp[4];
    }
#pragma unroll
    for (int off = 16; off > 0; off >>= 1) {
        a1s += __shfl_xor(a1s, off);
        z0  += __shfl_xor(z0, off);
        z1  += __shfl_xor(z1, off);
        z2  += __shfl_xor(z2, off);
        z3  += __shfl_xor(z3, off);
        z4  += __shfl_xor(z4, off);
    }
    if (lane == 0) {
        float alpha = fmaxf(sigmoidf(a1s + b1[0]), 0.0f) * mcur;
        alphas_t[b * TT1 + t] = alpha + TAILC * (mprev - mcur);

        const float cb = vmat[DD * UU] + b2[0];   // up_b.w2 + b2
        float s = fmaxf(sigmoidf(z0 + cb), 0.0f)
                + fmaxf(sigmoidf(z1 + cb), 0.0f)
                + fmaxf(sigmoidf(z2 + cb), 0.0f)
                + fmaxf(sigmoidf(z3 + cb), 0.0f)
                + fmaxf(sigmoidf(z4 + cb), 0.0f);
        a2partial[b * TT + t] = s * mcur;
    }
}

// ---------------------------------------------------------------------------
// 6) Deterministic per-batch reductions: token_num = floor(sum alphas_t),
//    token_num2 = sum a2partial. Fixed-order tree => bit-stable across replays.
// ---------------------------------------------------------------------------
__global__ __launch_bounds__(256) void reduce_kernel(
    const float* __restrict__ alphas_t, const float* __restrict__ a2partial,
    float* __restrict__ token_num, float* __restrict__ token_num2)
{
    __shared__ float s[256];
    const int b = blockIdx.x, tid = threadIdx.x;

    float acc = 0.0f;
    for (int t = tid; t < TT1; t += 256) acc += alphas_t[b * TT1 + t];
    s[tid] = acc; __syncthreads();
    for (int k = 128; k > 0; k >>= 1) {
        if (tid < k) s[tid] += s[tid + k];
        __syncthreads();
    }
    if (tid == 0) token_num[b] = floorf(s[0]);
    __syncthreads();

    acc = 0.0f;
    for (int t = tid; t < TT; t += 256) acc += a2partial[b * TT + t];
    s[tid] = acc; __syncthreads();
    for (int k = 128; k > 0; k >>= 1) {
        if (tid < k) s[tid] += s[tid + k];
        __syncthreads();
    }
    if (tid == 0) token_num2[b] = s[0];
}

// ---------------------------------------------------------------------------
// 7) CIF scalar scan: 1 thread per batch. Emits cif_peak, per-step segment
//    weights (w_in / w_next), fire positions and fire counts.
// ---------------------------------------------------------------------------
__global__ void cif_scan_kernel(
    const float* __restrict__ alphas_t, float* __restrict__ cif_peak,
    float* __restrict__ w_in, float* __restrict__ w_next,
    int* __restrict__ fpos, int* __restrict__ nf)
{
    const int b = threadIdx.x;
    if (b >= BB) return;
    float integ = 0.0f;
    int cnt = 0;
    for (int t = 0; t < TT1; ++t) {
        float alpha = alphas_t[b * TT1 + t];
        float dist  = 1.0f - integ;
        integ += alpha;
        cif_peak[b * TT1 + t] = integ;          // fires_t (pre-subtraction)
        bool fire = integ >= 1.0f;
        float cur = fire ? dist : alpha;
        w_in[b * TT1 + t]   = cur;              // weight into current segment
        w_next[b * TT1 + t] = fire ? (alpha - cur) : 0.0f; // leftover -> next seg
        if (fire) { integ -= 1.0f; fpos[b * 1008 + cnt] = t; ++cnt; }
    }
    nf[b] = cnt;
}

// ---------------------------------------------------------------------------
// 8) Acoustic gather: one block per (segment p, batch b). Each output row is a
//    short weighted sum of encoder rows over its fire segment; zeros beyond nf.
// ---------------------------------------------------------------------------
__global__ __launch_bounds__(256) void acoustic_kernel(
    const float* __restrict__ enc, const float* __restrict__ w_in,
    const float* __restrict__ w_next, const int* __restrict__ fpos,
    const int* __restrict__ nf, float* __restrict__ acoustic)
{
    const int p = blockIdx.x;
    const int b = blockIdx.y;
    const int d = threadIdx.x;

    float a0 = 0.0f, a1 = 0.0f;
    const int n = nf[b];
    if (p < n) {
        const int ts   = (p == 0) ? 0 : fpos[b * 1008 + p - 1];
        const int te   = fpos[b * 1008 + p];
        const int tbeg = (p == 0) ? 0 : ts + 1;
        if (p > 0 && ts < TT) {
            const float w = w_next[b * TT1 + ts];
            const float* h = enc + ((((long)b * TT) + ts) << 9);
            a0 += w * h[d]; a1 += w * h[d + 256];
        }
        for (int t = tbeg; t <= te; ++t) {
            if (t < TT) {
                const float w = w_in[b * TT1 + t];
                const float* h = enc + ((((long)b * TT) + t) << 9);
                a0 += w * h[d]; a1 += w * h[d + 256];
            }
        }
    }
    float* out = acoustic + ((((long)b * TT1) + p) << 9);
    out[d] = a0;
    out[d + 256] = a1;
}

// ---------------------------------------------------------------------------
extern "C" void kernel_launch(void* const* d_in, const int* in_sizes, int n_in,
                              void* d_out, int out_size, void* d_ws, size_t ws_size,
                              hipStream_t stream) {
    (void)in_sizes; (void)n_in; (void)out_size; (void)ws_size;

    const float* enc    = (const float*)d_in[0];
    const float* mask   = (const float*)d_in[1];
    const float* conv_w = (const float*)d_in[2];
    const float* conv_b = (const float*)d_in[3];
    const float* up_w   = (const float*)d_in[4];
    const float* up_b   = (const float*)d_in[5];
    const float* w1     = (const float*)d_in[6];
    const float* b1     = (const float*)d_in[7];
    const float* w2     = (const float*)d_in[8];
    const float* b2     = (const float*)d_in[9];

    // ---- workspace carve-out ----
    char* ws = (char*)d_ws;
    size_t off = 0;
    auto carve = [&](size_t bytes) -> char* {
        char* p = ws + off;
        off = (off + bytes + 255) & ~(size_t)255;
        return p;
    };
    unsigned short* Apad    = (unsigned short*)carve((size_t)BB * APAD_T * DD * 2);
    unsigned short* Wc      = (unsigned short*)carve((size_t)KC * DD * DD * 2);
    float*          outF    = (float*)carve((size_t)BB * TP * DD * 4);
    float*          vmat    = (float*)carve((size_t)(DD * UU + 1) * 4);
    float*          a2part  = (float*)carve((size_t)BB * TT * 4);
    float*          w_in    = (float*)carve((size_t)BB * TT1 * 4);
    float*          w_next  = (float*)carve((size_t)BB * TT1 * 4);
    int*            fpos    = (int*)carve((size_t)BB * 1008 * 4);
    int*            nf      = (int*)carve((size_t)BB * 4);

    // ---- output carve-out (reference return order, all fp32) ----
    float* out        = (float*)d_out;
    float* acoustic   = out;                                  // [B,1001,512]
    float* token_num  = acoustic + (size_t)BB * TT1 * DD;     // [B]
    float* alphas_t   = token_num + BB;                       // [B,1001]
    float* cif_peak   = alphas_t + (size_t)BB * TT1;          // [B,1001]
    float* token_num2 = cif_peak + (size_t)BB * TT1;          // [B]

    // 1) prep activations (pad + bf16 cast)
    {
        long n = (long)BB * APAD_T * DD;
        prep_apad_kernel<<<(unsigned)((n + 255) / 256), 256, 0, stream>>>(enc, Apad);
    }
    // 2) prep conv weights (repack + bf16 cast)
    prep_wc_kernel<<<(KC * DD * DD + 255) / 256, 256, 0, stream>>>(conv_w, Wc);
    // 3) fuse up_w with w2
    prep_v_kernel<<<(DD * UU + 1 + 255) / 256, 256, 0, stream>>>(up_w, up_b, w2, vmat);
    // 4) conv as WMMA GEMM with TDM->LDS pipeline (+bias+ReLU)
    conv_gemm_wmma_kernel<<<dim3(DD / 64, TP / 128, BB), 256, 0, stream>>>(
        Apad, Wc, conv_b, outF);
    // 5) pointwise dots: alphas_t (with tail-add) + alphas2 partials
    pointwise_kernel<<<(BB * TT1 + 7) / 8, 256, 0, stream>>>(
        outF, mask, vmat, w1, b1, b2, alphas_t, a2part);
    // 6) deterministic reductions -> token_num, token_num2
    reduce_kernel<<<BB, 256, 0, stream>>>(alphas_t, a2part, token_num, token_num2);
    // 7) CIF scalar scan -> cif_peak + segment metadata
    cif_scan_kernel<<<1, 32, 0, stream>>>(alphas_t, cif_peak, w_in, w_next, fpos, nf);
    // 8) parallel segment gather -> acoustic_embeds (zeros past last fire)
    acoustic_kernel<<<dim3(TT1, BB), 256, 0, stream>>>(
        enc, w_in, w_next, fpos, nf, acoustic);
}
